// SimpleRNN_45844480918324
// MI455X (gfx1250) — matrix operand order repaired and checked by reading, not compiled, not run on hardware
//
#include <hip/hip_runtime.h>

// SimpleRNN scan: B=1024, T=2048, I=1, H=64, O=1.
// One wave (32 lanes) owns a 16-batch tile and runs the full T=2048 scan with
// h and W_h resident in registers. Recurrent matmul: Hnew^T = W_h x H^T via
// v_wmma_f32_16x16x32_f16 (A = W_h fragments, loop-invariant; B = h in f16).
// A static hidden-index permutation makes the f32 D layout identical to the
// next step's f16 B layout, so the only per-step conversion is v_cvt_pk.

typedef _Float16 v16h __attribute__((ext_vector_type(16)));
typedef __fp16   v2hf __attribute__((ext_vector_type(2)));
typedef float    v8f  __attribute__((ext_vector_type(8)));

#define RNN_B 1024
#define RNN_T 2048
#define RNN_H 64

union H16 { v16h v; v2hf p[8]; };

__device__ __forceinline__ float fast_tanh(float x) {
#if __has_builtin(__builtin_amdgcn_tanhf)
  return __builtin_amdgcn_tanhf(x);          // gfx1250 V_TANH_F32 (trans op)
#else
  x = fminf(fmaxf(x, -9.0f), 9.0f);
  float e = __expf(2.0f * x);
  return (e - 1.0f) / (e + 1.0f);
#endif
}

// add value from lane^16 (ds_swizzle SWAPX16: xor=0x10, or=0, and=0x1f)
__device__ __forceinline__ float xor16_sum(float v) {
  int o = __builtin_amdgcn_ds_swizzle(__float_as_int(v), 0x401F);
  return v + __int_as_float(o);
}

__global__ __launch_bounds__(32, 1)
void rnn_wmma_kernel(const float* __restrict__ x_seq,
                     const float* __restrict__ W_h,
                     const float* __restrict__ W_x,
                     const float* __restrict__ W_y,
                     float* __restrict__ out) {
  const int lane = threadIdx.x & 31;
  const int sl   = lane >> 4;    // lane half: selects K sub-range (A) / M+8 (D)
  const int m    = lane & 15;    // A-row within tile; batch column index
  const int b    = blockIdx.x * 16 + m;

  // ---- Load A = W_h (rows permuted per hidden-index map) into f16 A-layout.
  // A (16x32 f16), contiguous k order used here, same as B: lanes 0-15 row M=lane, halves 0..15 -> K=0..15;
  // lanes 16-31 same rows, K=16..31.
  v16h A[4][2];
#pragma unroll
  for (int jt = 0; jt < 4; ++jt) {
    const int jphys = 32 * (jt >> 1) + 16 * (m >> 3) + 8 * (jt & 1) + (m & 7);
    const float* wrow = W_h + jphys * RNN_H;
#pragma unroll
    for (int c = 0; c < 2; ++c) {
      H16 u;
#pragma unroll
      for (int i = 0; i < 8; ++i) {
        const int hh0 = 2 * i, hh1 = 2 * i + 1;
        const int k0 = 32 * c + 16 * sl + hh0;
        const int k1 = 32 * c + 16 * sl + hh1;
        v2hf pr;
        pr[0] = (__fp16)wrow[k0];
        pr[1] = (__fp16)wrow[k1];
        u.p[i] = pr;
      }
      A[jt][c] = u.v;
    }
  }

  // ---- Per-lane W_x / W_y gathered in the D-tile element order.
  // D tile jt, acc element r, lane half sl holds hidden unit:
  //   j = 32*(jt>>1) + 16*sl + 8*(jt&1) + r
  float wx[4][8], wy[4][8];
#pragma unroll
  for (int jt = 0; jt < 4; ++jt)
#pragma unroll
    for (int r = 0; r < 8; ++r) {
      const int j = 32 * (jt >> 1) + 16 * sl + 8 * (jt & 1) + r;
      wx[jt][r] = W_x[j];
      wy[jt][r] = W_y[j];
    }

  const float* xrow = x_seq + (size_t)b * RNN_T;
  float*       orow = out   + (size_t)b * RNN_T;

  v16h B0 = {};  // h (f16, B-operand layout, K=0..31)  -- h0 = 0
  v16h B1 = {};  // h (f16, B-operand layout, K=32..63)

  for (int t0 = 0; t0 < RNN_T; t0 += 4) {
    const float4 xv4 = *reinterpret_cast<const float4*>(xrow + t0);
    const float xs[4] = {xv4.x, xv4.y, xv4.z, xv4.w};
    float yl[4];

#pragma unroll
    for (int u = 0; u < 4; ++u) {
      // acc init = x_t * W_x bias (outer product), in D layout
      v8f acc[4];
#pragma unroll
      for (int jt = 0; jt < 4; ++jt)
#pragma unroll
        for (int r = 0; r < 8; ++r) acc[jt][r] = xs[u] * wx[jt][r];

      // pre = W_h x h^T + bias : two chained K=32 WMMAs per 16x16 tile
#pragma unroll
      for (int jt = 0; jt < 4; ++jt) {
        acc[jt] = __builtin_amdgcn_wmma_f32_16x16x32_f16(
            false, A[jt][0], false, B0, (short)0, acc[jt], false, false);
        acc[jt] = __builtin_amdgcn_wmma_f32_16x16x32_f16(
            false, A[jt][1], false, B1, (short)0, acc[jt], false, false);
      }

      // h = tanh(pre); y partial = wy.h; repack h into next-step B operands.
      // D tile 0 -> B0 halves 0..7, tile 1 -> B0 halves 8..15,
      // tile 2 -> B1 halves 0..7, tile 3 -> B1 halves 8..15 (per-lane packs).
      H16 nB0, nB1;
      float yp = 0.0f;
#pragma unroll
      for (int jt = 0; jt < 4; ++jt) {
#pragma unroll
        for (int i = 0; i < 4; ++i) {
          const float a  = fast_tanh(acc[jt][2 * i]);
          const float c2 = fast_tanh(acc[jt][2 * i + 1]);
          yp += wy[jt][2 * i] * a + wy[jt][2 * i + 1] * c2;
          const v2hf pr = __builtin_amdgcn_cvt_pkrtz(a, c2);
          if (jt == 0)      nB0.p[i]     = pr;
          else if (jt == 1) nB0.p[4 + i] = pr;
          else if (jt == 2) nB1.p[i]     = pr;
          else              nB1.p[4 + i] = pr;
        }
      }
      B0 = nB0.v;
      B1 = nB1.v;

      // y[b] = partial(lane) + partial(lane^16); off the recurrence path
      yl[u] = xor16_sum(yp);
    }

    if (lane < 16) {
      float4 yv;
      yv.x = yl[0]; yv.y = yl[1]; yv.z = yl[2]; yv.w = yl[3];
      *reinterpret_cast<float4*>(orow + t0) = yv;
    }
  }
}

extern "C" void kernel_launch(void* const* d_in, const int* in_sizes, int n_in,
                              void* d_out, int out_size, void* d_ws, size_t ws_size,
                              hipStream_t stream) {
  const float* x  = (const float*)d_in[0];  // x_seq [1024,2048,1]
  const float* Wh = (const float*)d_in[1];  // W_h   [64,64]
  const float* Wx = (const float*)d_in[2];  // W_x   [64,1]
  const float* Wy = (const float*)d_in[3];  // W_y   [1,64]
  float* out = (float*)d_out;               // [1024,2048,1]

  dim3 grid(RNN_B / 16);  // 64 waves, one 16-batch tile each
  dim3 block(32);
  hipLaunchKernelGGL(rnn_wmma_kernel, grid, block, 0, stream, x, Wh, Wx, Wy, out);
}
